// SlayerSNN_8787503088254
// MI455X (gfx1250) — matrix operand order repaired and checked by reading, not compiled, not run on hardware
//
#include <hip/hip_runtime.h>

typedef __attribute__((ext_vector_type(16))) _Float16 v16h;
typedef __attribute__((ext_vector_type(8)))  _Float16 v8h;
typedef __attribute__((ext_vector_type(4)))  _Float16 v4h;
typedef __attribute__((ext_vector_type(8)))  float    v8f;
typedef __attribute__((ext_vector_type(4)))  float    v4f;
typedef __attribute__((ext_vector_type(4)))  int      v4i;

#define T_LEN  100
#define KPAD_T 128          // K dim of psp GEMM padded to 4 wmma k-tiles
#define NPAD_T 112          // N (time) dim padded to 7 wmma n-tiles
#define THETA 10.0f

// Build v16h fragment from two contiguous 8-half LDS runs (2x ds_load_b128).
__device__ __forceinline__ v16h frag_a(const _Float16* base, int lo_off, int hi_off) {
    v8h alo = *(const v8h*)(base + lo_off);
    v8h ahi = *(const v8h*)(base + hi_off);
    return __builtin_shufflevector(alo, ahi, 0,1,2,3,4,5,6,7,8,9,10,11,12,13,14,15);
}

// ---------------------------------------------------------------------------
// KT transposed: KTt[t][k] = alpha(t-k), t in [0,112), k in [0,128).
// alpha(d) = (d/10)*exp(1 - d/10) for 0 <= d, valid k<100, t<100.
// ---------------------------------------------------------------------------
__global__ void init_kt_kernel(_Float16* __restrict__ kt) {
    int i = blockIdx.x * blockDim.x + threadIdx.x;
    if (i >= NPAD_T * KPAD_T) return;
    int t = i / KPAD_T, k = i % KPAD_T;
    float v = 0.f;
    if (k < T_LEN && t < T_LEN && t >= k) {
        float d = (float)(t - k);
        v = d * 0.1f * __expf(1.0f - d * 0.1f);
    }
    kt[i] = (_Float16)v;
}

// ---------------------------------------------------------------------------
// psp as GEMM: Y[R x 100] = X[R x 100] * KT.  4 waves/block, one 16-row
// stripe per wave.  KTt (transposed) staged in LDS once per block.
// ---------------------------------------------------------------------------
__global__ __launch_bounds__(128) void psp_gemm_kernel(
        const float* __restrict__ x, float* __restrict__ y,
        int R, const _Float16* __restrict__ ktg) {
    __shared__ _Float16 KTt[NPAD_T * KPAD_T];  // 28672 B, [t][k]
    __shared__ _Float16 XS[4][16 * KPAD_T];    // 4 x 4096 B, [row][k]
    const int tid  = threadIdx.x;
    const int wave = tid >> 5, lane = tid & 31;

    {   // 16-byte cooperative copy of KTt
        const v4i* src = (const v4i*)ktg;
        v4i*       dst = (v4i*)KTt;
        for (int i = tid; i < (NPAD_T * KPAD_T) / 8; i += 128) dst[i] = src[i];
    }

    const int nStripes = R >> 4;
    const int stripe   = blockIdx.x * 4 + wave;
    if (stripe < nStripes) {
        const int r0 = stripe * 16;
        __builtin_prefetch(&x[(size_t)r0 * T_LEN], 0, 1);   // global_prefetch_b8
        // lane owns column-group (lane*4); rows 0..15.  25 groups hold data.
        const int c4 = lane * 4;
#pragma unroll
        for (int row = 0; row < 16; ++row) {
            v4h hv = {};
            if (c4 < T_LEN) {
                v4f f = *(const v4f*)(x + (size_t)(r0 + row) * T_LEN + c4);
                hv[0] = (_Float16)f[0]; hv[1] = (_Float16)f[1];
                hv[2] = (_Float16)f[2]; hv[3] = (_Float16)f[3];
            }
            *(v4h*)(&XS[wave][row * KPAD_T + c4]) = hv;     // ds_store_b64
        }
    }
    __syncthreads();
    if (stripe >= nStripes) return;          // wave-uniform; EXEC stays full

    const int r0   = stripe * 16;
    const int lrow = lane & 15;
    const int half = lane >> 4;
    const _Float16* xs = &XS[wave][0];

#pragma unroll
    for (int nt = 0; nt < NPAD_T / 16; ++nt) {
        const int t0 = nt * 16;
        v8f acc = {};
#pragma unroll
        for (int kt = 0; kt < KPAD_T / 32; ++kt) {
            const int k0 = kt * 32;
            v16h a = frag_a(xs, lrow * KPAD_T + k0 + 16 * half,
                                lrow * KPAD_T + k0 + 16 * half + 8);
            v16h b = *(const v16h*)(&KTt[(t0 + lrow) * KPAD_T + k0 + 16 * half]);
            acc = __builtin_amdgcn_wmma_f32_16x16x32_f16(
                      false, a, false, b, (short)0, acc, false, false);
        }
        const int t = t0 + lrow;
        if (t < T_LEN) {
#pragma unroll
            for (int i = 0; i < 8; ++i)      // lanes 0-15: 16 consecutive t -> 64B
                y[(size_t)(r0 + i + 8 * half) * T_LEN + t] = acc[i];
        }
    }
}

// ---------------------------------------------------------------------------
// conv_t as implicit GEMM, geometry as template params so the K loop fully
// unrolls.  One (n,t) per blockIdx.x, 16 out-channels per blockIdx.z,
// 4 pixel tiles (one per wave) per blockIdx.y.
// A = weights [o][k] (LDS, shared), B = im2col [pixel][k] (LDS, per wave).
// ---------------------------------------------------------------------------
template<int C, int H, int W, int O, int Ho, int Wo, int KH, int KW, int PAD>
__global__ __launch_bounds__(128) void conv_wmma_kernel(
        const float* __restrict__ in, const float* __restrict__ wgt,
        float* __restrict__ out) {
    constexpr int Klen = C * KH * KW;
    constexpr int Kpad = ((Klen + 31) / 32) * 32;
    constexpr int KHW  = KH * KW;
    constexpr int HWo  = Ho * Wo;
    __shared__ _Float16 AS[16 * Kpad];          // [o][k]
    __shared__ _Float16 BS[4][16 * Kpad];       // [pixel][k]
    const int tid  = threadIdx.x;
    const int wave = tid >> 5, lane = tid & 31;
    const int nt = blockIdx.x;
    const int n  = nt / T_LEN, t = nt % T_LEN;
    const int ob = blockIdx.z * 16;
    const int p0 = (blockIdx.y * 4 + wave) * 16;

    // stage weights: k contiguous across threads -> coalesced
#pragma unroll
    for (int o = 0; o < 16; ++o)
        for (int k = tid; k < Kpad; k += 128) {
            float v = (k < Klen) ? wgt[(size_t)(ob + o) * Klen + k] : 0.f;
            AS[o * Kpad + k] = (_Float16)v;
        }
    // stage im2col tile (transposed: [p][k]) -- gather reads, contiguous ds writes
    _Float16* Bw = &BS[wave][0];
#pragma unroll
    for (int p = 0; p < 16; ++p) {
        const int pix = p0 + p;
        const int oy = pix / Wo, ox = pix % Wo;
        for (int k = lane; k < Kpad; k += 32) {
            float v = 0.f;
            if (k < Klen) {
                int c = k / KHW, rem = k % KHW;
                int ky = rem / KW, kx = rem % KW;
                int iy = oy + ky - PAD, ix = ox + kx - PAD;
                if (iy >= 0 && iy < H && ix >= 0 && ix < W)
                    v = in[((size_t)((n * C + c) * H + iy) * W + ix) * T_LEN + t];
            }
            Bw[p * Kpad + k] = (_Float16)v;
        }
    }
    __syncthreads();

    const int lrow = lane & 15, half = lane >> 4;
    v8f acc = {};
#pragma unroll
    for (int kt = 0; kt < Kpad / 32; ++kt) {
        const int k0 = kt * 32;
        v16h a = frag_a(AS, lrow * Kpad + k0 + 16 * half,
                            lrow * Kpad + k0 + 16 * half + 8);
        v16h b = *(const v16h*)(&Bw[lrow * Kpad + k0 + 16 * half]);
        acc = __builtin_amdgcn_wmma_f32_16x16x32_f16(
                  false, a, false, b, (short)0, acc, false, false);
    }
#pragma unroll
    for (int i = 0; i < 8; ++i) {
        int o   = ob + i + 8 * half;
        int pix = p0 + lrow;
        out[((size_t)(n * O + o) * HWo + pix) * T_LEN + t] = acc[i];
    }
}

// ---------------------------------------------------------------------------
// fc: m6[o,(n,t)] = sum_k Wfc[o,k] * psp5[n,k,t].  M=16 (10 valid), K=4096,
// N=800.  One wave per block; A rows loaded as float4, B gathered (strided).
// ---------------------------------------------------------------------------
__global__ __launch_bounds__(32) void fc_wmma_kernel(
        const float* __restrict__ psp5, const float* __restrict__ wfc,
        float* __restrict__ m6) {
    const int lane = threadIdx.x & 31;
    const int col0 = blockIdx.x * 16;
    const int lrow = lane & 15, half = lane >> 4;
    const int ntc = col0 + lrow;
    const int n = ntc / T_LEN, t = ntc % T_LEN;
    const bool arow_ok = (lrow < 10);
    const float* wrow = wfc + (size_t)lrow * 4096;
    v8f acc = {};
    for (int kt = 0; kt < 4096 / 32; ++kt) {
        const int k0 = kt * 32;
        v16h a = {}, b = {};
        if (arow_ok) {
            v4f w0 = *(const v4f*)(wrow + k0 + 16 * half);
            v4f w1 = *(const v4f*)(wrow + k0 + 16 * half + 4);
            v4f w2 = *(const v4f*)(wrow + k0 + 16 * half + 8);
            v4f w3 = *(const v4f*)(wrow + k0 + 16 * half + 12);
#pragma unroll
            for (int e = 0; e < 4; ++e) {
                a[e]      = (_Float16)w0[e];
                a[4 + e]  = (_Float16)w1[e];
                a[8 + e]  = (_Float16)w2[e];
                a[12 + e] = (_Float16)w3[e];
            }
        }
#pragma unroll
        for (int j = 0; j < 16; ++j) {
            int kb = k0 + half * 16 + j;
            b[j] = (_Float16)psp5[((size_t)n * 4096 + kb) * T_LEN + t];
        }
        acc = __builtin_amdgcn_wmma_f32_16x16x32_f16(
                  false, a, false, b, (short)0, acc, false, false);
    }
#pragma unroll
    for (int i = 0; i < 8; ++i) {
        int o = i + 8 * half;
        if (o < 10)
            m6[((size_t)n * 10 + o) * T_LEN + t] = acc[i];
    }
}

// ---------------------------------------------------------------------------
// 2x2 sum-pool * 2.75 ; 4 time-steps per thread via float4.
// ---------------------------------------------------------------------------
__global__ void pool_kernel(const float* __restrict__ in, float* __restrict__ out,
                            int H, int W, int total4) {
    int idx = blockIdx.x * blockDim.x + threadIdx.x;
    if (idx >= total4) return;
    const int Wo = W >> 1, Ho = H >> 1;
    int t4 = (idx % 25) * 4; int r = idx / 25;
    int x = r % Wo; r /= Wo;
    int y = r % Ho; r /= Ho;          // r = n*C + c
    const float* base = in + ((size_t)(r * H + 2 * y) * W + 2 * x) * T_LEN + t4;
    v4f s = *(const v4f*)(base)
          + *(const v4f*)(base + T_LEN)
          + *(const v4f*)(base + (size_t)W * T_LEN)
          + *(const v4f*)(base + (size_t)(W + 1) * T_LEN);
    size_t o = ((size_t)(idx / 25) * T_LEN) + t4;
    *(v4f*)(out + o) = s * 2.75f;
}

// ---------------------------------------------------------------------------
// spike: sequential refractory scan, 15-tap correction in registers,
// 4 time-steps per float4 memory transaction.
// ---------------------------------------------------------------------------
__global__ void spike_kernel(const float* __restrict__ m, float* __restrict__ s,
                             int R) {
    int r = blockIdx.x * blockDim.x + threadIdx.x;
    if (r >= R) return;
    float rt[15];
#pragma unroll
    for (int j = 0; j < 15; ++j) {
        float tt = (float)(j + 1);
        rt[j] = -2.0f * THETA * tt * __expf(1.0f - tt);   // REF_KERNEL[j+1]
    }
    float corr[15];
#pragma unroll
    for (int j = 0; j < 15; ++j) corr[j] = 0.f;
    const float* mm = m + (size_t)r * T_LEN;
    float*       ss = s + (size_t)r * T_LEN;
    for (int tb = 0; tb < T_LEN / 4; ++tb) {
        v4f mv = *(const v4f*)(mm + tb * 4);
        v4f sv;
#pragma unroll
        for (int e = 0; e < 4; ++e) {
            float u  = mv[e] + corr[0];
            float sp = (u >= THETA) ? 1.0f : 0.0f;
#pragma unroll
            for (int j = 0; j < 14; ++j) corr[j] = corr[j + 1] + sp * rt[j];
            corr[14] = sp * rt[14];
            sv[e] = sp;
        }
        *(v4f*)(ss + tb * 4) = sv;
    }
}

// ---------------------------------------------------------------------------
extern "C" void kernel_launch(void* const* d_in, const int* in_sizes, int n_in,
                              void* d_out, int out_size, void* d_ws, size_t ws_size,
                              hipStream_t stream) {
    const float* x0  = (const float*)d_in[0];   // (8,3,34,34,100)
    const float* W1  = (const float*)d_in[1];   // (16,3,5,5)
    const float* W2  = (const float*)d_in[2];   // (32,16,3,3)
    const float* W3  = (const float*)d_in[3];   // (64,32,3,3)
    const float* Wfc = (const float*)d_in[4];   // (10,64,8,8)
    float* out = (float*)d_out;                 // (8,10,1,1,100) = 8000

    float*     ws  = (float*)d_ws;
    _Float16*  kt  = (_Float16*)ws;             // 112*128 halves = 7168 floats
    const size_t BUF = (size_t)8 * 16 * 32 * 32 * T_LEN;   // 13,107,200 floats
    float* bufA = ws + 8192;
    float* bufB = bufA + BUF;
    float* bufC = bufB + BUF;

    init_kt_kernel<<<(NPAD_T * KPAD_T + 255) / 256, 256, 0, stream>>>(kt);

    // ---- stage 1: s1 = spike(conv(psp(x), W1)) -----------------------------
    const int R0 = 8 * 3 * 34 * 34;             // 27744
    psp_gemm_kernel<<<(R0 / 16 + 3) / 4, 128, 0, stream>>>(x0, bufA, R0, kt);
    dim3 g1(8 * T_LEN, (32 * 32 / 16) / 4, 16 / 16);       // (800,16,1)
    conv_wmma_kernel<3,34,34,16,32,32,5,5,1><<<g1, 128, 0, stream>>>(bufA, W1, bufB);
    const int R1 = 8 * 16 * 32 * 32;            // 131072
    spike_kernel<<<(R1 + 255) / 256, 256, 0, stream>>>(bufB, bufC, R1);

    // ---- stage 2: s2 = spike(pool(psp(s1))) --------------------------------
    psp_gemm_kernel<<<(R1 / 16 + 3) / 4, 128, 0, stream>>>(bufC, bufA, R1, kt);
    const int R2 = 8 * 16 * 16 * 16;            // 32768
    pool_kernel<<<(R2 * 25 + 255) / 256, 256, 0, stream>>>(bufA, bufB, 32, 32, R2 * 25);
    spike_kernel<<<(R2 + 255) / 256, 256, 0, stream>>>(bufB, bufC, R2);

    // ---- stage 3: s3 = spike(conv(psp(s2), W2)) ----------------------------
    psp_gemm_kernel<<<(R2 / 16 + 3) / 4, 128, 0, stream>>>(bufC, bufA, R2, kt);
    dim3 g2(8 * T_LEN, (16 * 16 / 16) / 4, 32 / 16);       // (800,4,2)
    conv_wmma_kernel<16,16,16,32,16,16,3,3,1><<<g2, 128, 0, stream>>>(bufA, W2, bufB);
    const int R3 = 8 * 32 * 16 * 16;            // 65536
    spike_kernel<<<(R3 + 255) / 256, 256, 0, stream>>>(bufB, bufC, R3);

    // ---- stage 4: s4 = spike(pool(psp(s3))) --------------------------------
    psp_gemm_kernel<<<(R3 / 16 + 3) / 4, 128, 0, stream>>>(bufC, bufA, R3, kt);
    const int R4 = 8 * 32 * 8 * 8;              // 16384
    pool_kernel<<<(R4 * 25 + 255) / 256, 256, 0, stream>>>(bufA, bufB, 16, 16, R4 * 25);
    spike_kernel<<<(R4 + 255) / 256, 256, 0, stream>>>(bufB, bufC, R4);

    // ---- stage 5: s5 = spike(conv(psp(s4), W3)) ----------------------------
    psp_gemm_kernel<<<(R4 / 16 + 3) / 4, 128, 0, stream>>>(bufC, bufA, R4, kt);
    dim3 g3(8 * T_LEN, (8 * 8 / 16) / 4, 64 / 16);         // (800,1,4)
    conv_wmma_kernel<32,8,8,64,8,8,3,3,1><<<g3, 128, 0, stream>>>(bufA, W3, bufB);
    const int R5 = 8 * 64 * 8 * 8;              // 32768
    spike_kernel<<<(R5 + 255) / 256, 256, 0, stream>>>(bufB, bufC, R5);

    // ---- stage 6: out = spike(fc(psp(s5))) ---------------------------------
    psp_gemm_kernel<<<(R5 / 16 + 3) / 4, 128, 0, stream>>>(bufC, bufA, R5, kt);
    fc_wmma_kernel<<<(8 * T_LEN) / 16, 32, 0, stream>>>(bufA, Wfc, bufB);
    spike_kernel<<<(80 + 255) / 256, 256, 0, stream>>>(bufB, out, 80);
}
